// GFlowNet_34703335752333
// MI455X (gfx1250) — matrix operand order repaired and checked by reading, not compiled, not run on hardware
//
#include <hip/hip_runtime.h>

// ---------------------------------------------------------------------------
// GFlowNet TB-loss forward for MI455X (gfx1250, wave32, WMMA bf16).
// q2 network is dead code in the reference loss -> eliminated.
// Each block: 16 batch rows, 256 threads (8 waves), runs all 128 steps +
// final eval locally. GEMMs via v_wmma_f32_16x16x32_bf16; weights streamed
// from L2 with global_load_b128 (addrspace(1) pointers); biases staged to
// LDS via global_load_async_to_lds (ASYNCcnt) when the builtin exists.
// ---------------------------------------------------------------------------

typedef __attribute__((ext_vector_type(16))) __bf16 v16bf;
typedef __attribute__((ext_vector_type(8)))  float  v8f;
typedef __attribute__((ext_vector_type(4)))  unsigned u32x4;

#define AS1 __attribute__((address_space(1)))
#define AS3 __attribute__((address_space(3)))

#define VD    128
#define HD    512
#define BATCH 8192
#define MTILE 16
#define NTHREADS 256
#define LDA   544   // ushort (bf16) stride of A LDS buffer (16B-aligned rows)
#define LDH   520   // float stride of H LDS buffer

#if defined(__HIP_DEVICE_COMPILE__) && __has_builtin(__builtin_amdgcn_global_load_async_to_lds_b32)
#define USE_ASYNC_LDS 1
#else
#define USE_ASYNC_LDS 0
#endif

__device__ __forceinline__ unsigned short f2bf(float x) {
  unsigned u = __float_as_uint(x);
  unsigned r = (u + 0x7FFFu + ((u >> 16) & 1u)) >> 16;   // RNE
  return (unsigned short)r;
}

__device__ __forceinline__ const AS1 u32x4* gvec(const unsigned short* p) {
  return (const AS1 u32x4*)(const AS1 void*)p;   // generic -> global addrspace
}

__device__ __forceinline__ v16bf make_bfrag(u32x4 lo, u32x4 hi) {
  union { struct { u32x4 a, b; } p; v16bf v; } u;
  u.p.a = lo; u.p.b = hi;
  return u.v;
}

__device__ __forceinline__ v16bf load_afrag(const unsigned short* lo,
                                            const unsigned short* hi) {
  v16bf r;
  __builtin_memcpy(&r, lo, 16);                                   // ds_load_b128
  __builtin_memcpy(reinterpret_cast<char*>(&r) + 16, hi, 16);     // ds_load_b128
  return r;
}

// One wave computes NTW 16x16 output tiles of H[16xN] = A[16xK] * W^T.
// Register-blocked: one A fragment per K chunk feeds NTW WMMAs.
// Wt stored transposed [N][K] (bf16 bits) -> per-lane contiguous 32B loads.
template <int NTW>
__device__ __forceinline__ void wave_gemm(const unsigned short* __restrict__ Wt,
                                          const unsigned short* Albs,
                                          float* Hlds, int K) {
  const int lane = threadIdx.x & 31;
  const int wave = threadIdx.x >> 5;
  const int m    = lane & 15;
  const int g    = lane >> 4;          // K-half select (contiguous k order used here: k = 16*g + s)
  const unsigned short* arow = Albs + m * LDA;

  const unsigned short* wrow[NTW];
#pragma unroll
  for (int i = 0; i < NTW; ++i) {
    const int n = (wave + i * 8) * 16 + m;
    wrow[i] = Wt + (size_t)n * K + g * 16;
  }

  v8f acc[NTW];
#pragma unroll
  for (int i = 0; i < NTW; ++i) acc[i] = (v8f){};

  for (int kc = 0; kc < K; kc += 32) {
    if (kc + 32 < K) __builtin_prefetch(wrow[0] + kc + 32, 0, 1);
    v16bf a = load_afrag(arow + kc + g * 16, arow + kc + g * 16 + 8);
#pragma unroll
    for (int i = 0; i < NTW; ++i) {
      const AS1 u32x4* p = gvec(wrow[i] + kc);    // global_load_b128 x2
      v16bf b = make_bfrag(p[0], p[1]);
      acc[i] = __builtin_amdgcn_wmma_f32_16x16x32_bf16(
          false, a, false, b, (short)0, acc[i], false, false);
    }
  }
  // C layout: lanes 0-15 -> M = r, lanes 16-31 -> M = 8 + r ; N = lane&15
#pragma unroll
  for (int i = 0; i < NTW; ++i) {
    float* hp = Hlds + (wave + i * 8) * 16 + m;
#pragma unroll
    for (int r = 0; r < 8; ++r) hp[(g * 8 + r) * LDH] = acc[i][r];
  }
}

// LayerNorm(H + bias) * gamma + beta, ReLU, write bf16 into A buffer.
// 16 threads per row, 32 elements each (N = 512).
__device__ __forceinline__ void ln_relu(const float* __restrict__ bias,
                                        const float* __restrict__ gamma,
                                        const float* __restrict__ beta,
                                        float* Hlds, unsigned short* Albs,
                                        float* red) {
  const int tid = threadIdx.x;
  const int row = tid >> 4;
  const int c   = tid & 15;
  float s = 0.f, ss = 0.f;
#pragma unroll 4
  for (int j = 0; j < 32; ++j) {
    const int n = c * 32 + j;
    float x = Hlds[row * LDH + n] + bias[n];
    s += x; ss += x * x;
  }
  red[(row * 16 + c) * 2 + 0] = s;
  red[(row * 16 + c) * 2 + 1] = ss;
  __syncthreads();
  float S = 0.f, SS = 0.f;
#pragma unroll
  for (int j = 0; j < 16; ++j) {
    S  += red[(row * 16 + j) * 2 + 0];
    SS += red[(row * 16 + j) * 2 + 1];
  }
  const float mean = S * (1.0f / 512.0f);
  const float var  = SS * (1.0f / 512.0f) - mean * mean;
  const float rstd = rsqrtf(var + 1e-5f);
#pragma unroll 4
  for (int j = 0; j < 32; ++j) {
    const int n = c * 32 + j;
    float x = Hlds[row * LDH + n] + bias[n];
    float y = (x - mean) * rstd * gamma[n] + beta[n];
    Albs[row * LDA + n] = f2bf(fmaxf(y, 0.0f));
  }
  __syncthreads();
}

// Full q1 forward for the 16-row tile. A buffer must hold the input (bf16,
// first 128 cols). Leaves 384 logits per row in Hlds (bias added).
__device__ void qnet_forward(const unsigned short* WinT,
                             const unsigned short* WhT,
                             const unsigned short* WoutT,
                             const float* bin, const float* bh,
                             const float* lg, const float* lb,
                             const float* bout,
                             unsigned short* Albs, float* Hlds, float* red) {
  wave_gemm<4>(WinT, Albs, Hlds, VD);
  __syncthreads();
  ln_relu(bin, lg + 0 * HD, lb + 0 * HD, Hlds, Albs, red);

  wave_gemm<4>(WhT, Albs, Hlds, HD);
  __syncthreads();
  ln_relu(bh + 0 * HD, lg + 1 * HD, lb + 1 * HD, Hlds, Albs, red);

  wave_gemm<4>(WhT + (size_t)HD * HD, Albs, Hlds, HD);
  __syncthreads();
  ln_relu(bh + 1 * HD, lg + 2 * HD, lb + 2 * HD, Hlds, Albs, red);

  wave_gemm<3>(WoutT, Albs, Hlds, HD);   // 24 N-tiles
  __syncthreads();
  for (int i = threadIdx.x; i < MTILE * 3 * VD; i += NTHREADS) {
    const int r = i / (3 * VD), n = i % (3 * VD);
    Hlds[r * LDH + n] += bout[n];
  }
  __syncthreads();
}

__device__ __forceinline__ float rng01(unsigned row, unsigned t) {
  unsigned s = row * 2654435761u ^ (t * 0x9E3779B9u) ^ 0x85EBCA6Bu;
  s ^= s >> 16; s *= 0x7FEB352Du;
  s ^= s >> 15; s *= 0x846CA68Bu;
  s ^= s >> 16;
  return (float)(s >> 8) * (1.0f / 16777216.0f);
}

// Stage a small f32 array global -> LDS (async-to-LDS on gfx1250 if available).
__device__ __forceinline__ void stage_params(const float* g, float* l, int n) {
#if USE_ASYNC_LDS
  for (int i = threadIdx.x; i < n; i += NTHREADS) {
    __builtin_amdgcn_global_load_async_to_lds_b32(
        (AS1 int*)(AS1 void*)(void*)(g + i),
        (AS3 int*)(AS3 void*)(void*)(l + i), 0, 0);
  }
#else
  for (int i = threadIdx.x; i < n; i += NTHREADS) l[i] = g[i];
#endif
}

__global__ __launch_bounds__(NTHREADS) void gfn_main(
    const unsigned short* __restrict__ WinT,
    const unsigned short* __restrict__ WhT,
    const unsigned short* __restrict__ WoutT,
    const float* __restrict__ bin, const float* __restrict__ bh,
    const float* __restrict__ lg, const float* __restrict__ lb,
    const float* __restrict__ bout,
    const float* __restrict__ logZ, const float* __restrict__ energy_w,
    float* __restrict__ acc) {
  __shared__ signed char    Vst[MTILE][VD];     // state in {-1,0,+1}
  __shared__ unsigned short Abuf[MTILE][LDA];
  __shared__ float          Hbuf[MTILE][LDH];
  __shared__ float          red[MTILE * 16 * 2];
  __shared__ float          cbin[HD];           // LDS-resident biases
  __shared__ float          cbh[2 * HD];
  __shared__ float          cbout[3 * VD];
  __shared__ float          pf_sum[MTILE], pb_sum[MTILE];
  __shared__ int            prevloc[MTILE];

  const int tid = threadIdx.x;

  stage_params(bin, cbin, HD);
  stage_params(bh, cbh, 2 * HD);
  stage_params(bout, cbout, 3 * VD);
#if USE_ASYNC_LDS
#if __has_builtin(__builtin_amdgcn_s_wait_asynccnt)
  __builtin_amdgcn_s_wait_asynccnt(0);
#else
  asm volatile("s_wait_asynccnt 0" ::: "memory");
#endif
#endif

  for (int i = tid; i < MTILE * VD; i += NTHREADS) (&Vst[0][0])[i] = 0;
  if (tid < MTILE) { pf_sum[tid] = 0.f; pb_sum[tid] = 0.f; prevloc[tid] = 0; }
  __syncthreads();

  for (int t = 0; t < VD; ++t) {
    // stage V tile as bf16 into A buffer ({-1,0,+1} -> exact bf16 patterns)
    for (int i = tid; i < MTILE * VD; i += NTHREADS) {
      const int r = i >> 7, c = i & (VD - 1);
      const signed char v = Vst[r][c];
      Abuf[r][c] = (v == 0) ? (unsigned short)0
                            : (v > 0 ? (unsigned short)0x3F80u
                                     : (unsigned short)0xBF80u);
    }
    __syncthreads();

    qnet_forward(WinT, WhT, WoutT, cbin, cbh, lg, lb, cbout,
                 &Abuf[0][0], &Hbuf[0][0], red);

    if (tid < MTILE) {
      const int row = tid;
      const unsigned grow = blockIdx.x * MTILE + row;
      const float* lrow = &Hbuf[row][0];
      // ---- P_B for prev location (state BEFORE update); dropped at t==0 ----
      if (t > 0) {
        float mb = -3.0e38f;
        for (int j = 0; j < VD; ++j) {
          float v = lrow[2 * VD + j] - ((Vst[row][j] == 0) ? 1e20f : 0.f);
          mb = fmaxf(mb, v);
        }
        float Zb = 0.f;
        for (int j = 0; j < VD; ++j) {
          float v = lrow[2 * VD + j] - ((Vst[row][j] == 0) ? 1e20f : 0.f);
          Zb += __expf(v - mb);
        }
        const int p = prevloc[row];
        float dv = lrow[2 * VD + p] - ((Vst[row][p] == 0) ? 1e20f : 0.f);
        pb_sum[row] += dv - mb - __logf(Zb);
      }
      // ---- forward sample over 256 (loc,val) pairs ----
      float mx = -3.0e38f;
      for (int s = 0; s < 2 * VD; ++s) {
        float v = lrow[s] - ((Vst[row][s >> 1] != 0) ? 1e20f : 0.f);
        mx = fmaxf(mx, v);
      }
      float Z = 0.f;
      for (int s = 0; s < 2 * VD; ++s) {
        float v = lrow[s] - ((Vst[row][s >> 1] != 0) ? 1e20f : 0.f);
        Z += __expf(v - mx);
      }
      const float u = rng01(grow, (unsigned)t) * Z;
      float csum = 0.f, vlast = 0.f, ml_s = 0.f;
      int sample = -1;
      for (int s = 0; s < 2 * VD; ++s) {
        float v = lrow[s] - ((Vst[row][s >> 1] != 0) ? 1e20f : 0.f);
        csum += __expf(v - mx);
        vlast = v;
        if (sample < 0 && csum >= u) { sample = s; ml_s = v; }
      }
      if (sample < 0) { sample = 2 * VD - 1; ml_s = vlast; }
      pf_sum[row] += ml_s - mx - __logf(Z);
      const int loc = sample >> 1;
      Vst[row][loc] = (sample & 1) ? (signed char)1 : (signed char)-1;
      prevloc[row]  = loc;
    }
    __syncthreads();
  }

  // ---- final eval on fully-assigned state for pb_last ----
  for (int i = tid; i < MTILE * VD; i += NTHREADS) {
    const int r = i >> 7, c = i & (VD - 1);
    const signed char v = Vst[r][c];
    Abuf[r][c] = (v == 0) ? (unsigned short)0
                          : (v > 0 ? (unsigned short)0x3F80u
                                   : (unsigned short)0xBF80u);
  }
  __syncthreads();
  qnet_forward(WinT, WhT, WoutT, cbin, cbh, lg, lb, cbout,
               &Abuf[0][0], &Hbuf[0][0], red);

  if (tid < MTILE) {
    const int row = tid;
    const float* lrow = &Hbuf[row][0];
    float mb = -3.0e38f;
    for (int j = 0; j < VD; ++j) {
      float v = lrow[2 * VD + j] - ((Vst[row][j] == 0) ? 1e20f : 0.f);
      mb = fmaxf(mb, v);
    }
    float Zb = 0.f;
    for (int j = 0; j < VD; ++j) {
      float v = lrow[2 * VD + j] - ((Vst[row][j] == 0) ? 1e20f : 0.f);
      Zb += __expf(v - mb);
    }
    const int p = prevloc[row];
    float dv = lrow[2 * VD + p] - ((Vst[row][p] == 0) ? 1e20f : 0.f);
    const float pb_last = dv - mb - __logf(Zb);

    float score = 0.f;
    for (int j = 0; j < VD; ++j) score += (float)Vst[row][j] * energy_w[j];
    score = -score;

    const float diff = logZ[0] + pf_sum[row] - score - (pb_sum[row] + pb_last);
    atomicAdd(acc, diff * diff);
  }
}

// ---------------- prep / finalize kernels ----------------

__global__ void zero_acc(float* acc) { acc[0] = 0.0f; }

// src[R][C] (f32) -> dst[C][R] (bf16 bits)
__global__ void transpose_to_bf16(const float* __restrict__ src,
                                  unsigned short* __restrict__ dst,
                                  int R, int C) {
  const int i = blockIdx.x * blockDim.x + threadIdx.x;
  if (i < R * C) {
    const int r = i / C, c = i % C;
    dst[(size_t)c * R + r] = f2bf(src[i]);
  }
}

__global__ void finalize(const float* __restrict__ acc, float* __restrict__ out) {
  out[0] = acc[0] * (1.0f / (float)BATCH);
}

// ---------------- launch ----------------

extern "C" void kernel_launch(void* const* d_in, const int* in_sizes, int n_in,
                              void* d_out, int out_size, void* d_ws, size_t ws_size,
                              hipStream_t stream) {
  const float* q1_Win  = (const float*)d_in[0];   // [128][512]
  const float* q1_bin  = (const float*)d_in[1];   // [512]
  const float* q1_Wh   = (const float*)d_in[2];   // [2][512][512]
  const float* q1_bh   = (const float*)d_in[3];   // [2][512]
  const float* q1_lg   = (const float*)d_in[4];   // [3][512]
  const float* q1_lb   = (const float*)d_in[5];   // [3][512]
  const float* q1_Wout = (const float*)d_in[6];   // [512][384]
  const float* q1_bout = (const float*)d_in[7];   // [384]
  // d_in[8..15]: q2 params -- dead code in the reference loss, skipped.
  const float* logZ     = (const float*)d_in[16];
  const float* energy_w = (const float*)d_in[17];

  char* ws = (char*)d_ws;
  float* acc = (float*)ws;                                    // 1 float (+pad)
  unsigned short* WinT  = (unsigned short*)(ws + 256);        // [512][128]
  unsigned short* WhT   = WinT + (size_t)HD * VD;             // [2][512][512]
  unsigned short* WoutT = WhT + (size_t)2 * HD * HD;          // [384][512]

  zero_acc<<<1, 1, 0, stream>>>(acc);

  transpose_to_bf16<<<(VD * HD + 255) / 256, 256, 0, stream>>>(q1_Win, WinT, VD, HD);
  transpose_to_bf16<<<(HD * HD + 255) / 256, 256, 0, stream>>>(q1_Wh, WhT, HD, HD);
  transpose_to_bf16<<<(HD * HD + 255) / 256, 256, 0, stream>>>(
      q1_Wh + (size_t)HD * HD, WhT + (size_t)HD * HD, HD, HD);
  transpose_to_bf16<<<(HD * 3 * VD + 255) / 256, 256, 0, stream>>>(
      q1_Wout, WoutT, HD, 3 * VD);

  gfn_main<<<BATCH / MTILE, NTHREADS, 0, stream>>>(
      WinT, WhT, WoutT, q1_bin, q1_bh, q1_lg, q1_lb, q1_bout,
      logZ, energy_w, acc);

  finalize<<<1, 1, 0, stream>>>(acc, (float*)d_out);
}
